// DiTSelfAttention_16252156248268
// MI455X (gfx1250) — matrix operand order repaired and checked by reading, not compiled, not run on hardware
//
#include <hip/hip_runtime.h>
#include <hip/hip_bf16.h>

typedef __attribute__((ext_vector_type(16))) _Float16 v16h;
typedef __attribute__((ext_vector_type(8)))  float    v8f;
typedef __attribute__((ext_vector_type(4)))  unsigned int v4u32;
typedef __attribute__((ext_vector_type(8)))  int      v8i32;
typedef __attribute__((ext_vector_type(4)))  int      v4i32;

#define CC 2048   // channels
#define SS 2048   // sequence
#define MM 4096   // B*S rows
#define NH 16     // heads
#define HD 128    // head dim

// ---------------- TDM: 2D tile load global -> LDS (ISA ch.8 D# layout) ----------------
// 16-bit elements; tile_w elems per row, tile_h rows, row stride in elems.
// Descriptor groups are wave-uniform -> SGPRs. Tracked by TENSORcnt.

__device__ __forceinline__ unsigned lds_off_u32(const void* p) {
  // LDS aperture: ADDR[63:32] selects the aperture, ADDR[31:0] is the LDS byte address.
  return (unsigned)(unsigned long long)p;
}

__device__ __forceinline__ void tdm_load_2d(unsigned lds_addr, const void* gptr,
                                            int tile_w, int tile_h, int row_stride) {
  const unsigned long long ga = (unsigned long long)gptr;
  v4u32 g0;
  g0[0] = 1u;                                         // count=1, user mode, no gather
  g0[1] = lds_addr;                                   // lds_addr (bits 63:32 of D# group0)
  g0[2] = (unsigned)(ga & 0xffffffffu);               // global_addr[31:0]
  g0[3] = (unsigned)((ga >> 32) & 0x1ffffffu)         // global_addr[56:32]
        | (2u << 30);                                 // type = 2 ("image")
  v8i32 g1;
  g1[0] = (1 << 16);                                  // data_size=1 (2 bytes)
  g1[1] = (tile_w & 0xffff) << 16;                    // tensor_dim0[15:0] at bits 63:48
  g1[2] = (tile_h & 0xffff) << 16;                    // tensor_dim1[15:0] at bits 95:80
  g1[3] = (tile_w & 0xffff) << 16;                    // tile_dim0 at bits 127:112
  g1[4] = (tile_h & 0xffff);                          // tile_dim1 at bits 143:128
  g1[5] = row_stride;                                 // tensor_dim0_stride[31:0]
  g1[6] = 0;
  g1[7] = 0;
  const v4i32 z4 = {0, 0, 0, 0};                      // groups 2/3: NULL (2-D tensor)
  const v8i32 z8 = {0, 0, 0, 0, 0, 0, 0, 0};
  __builtin_amdgcn_tensor_load_to_lds(g0, g1, z4, z4, z8, 0);
}

// ---------------- WMMA fragment loaders (ISA 7.12.2 layouts, wave32) ----------------

__device__ __forceinline__ v16h frag_a_lds(const _Float16* p0, int stride, int lane) {
  const int r  = lane & 15;
  const int kb = (lane >> 4) << 3;              // 0 or 8
  const _Float16* p = p0 + r * stride + kb;
  v16h a;
#pragma unroll
  for (int i = 0; i < 8; ++i) { a[i] = p[i]; a[i + 8] = p[i + 16]; }
  return a;
}

__device__ __forceinline__ v16h frag_b_lds(const _Float16* p0, int stride, int lane) {
  const int nn = lane & 15;
  const int kb = (lane >> 4) << 3;              // 0 or 8
  const _Float16* p = p0 + nn * stride + kb;
  v16h b;
#pragma unroll
  for (int i = 0; i < 8; ++i) { b[i] = p[i]; b[i + 8] = p[i + 16]; }
  return b;
}

__device__ __forceinline__ v16h frag_b_rowmajor(const _Float16* p0, int stride, int lane) {
  const int nn = lane & 15;
  const int kb = (lane >> 4) << 3;
  v16h b;
#pragma unroll
  for (int i = 0; i < 8; ++i) { b[i] = p0[(kb + i) * stride + nn]; b[i + 8] = p0[(kb + 16 + i) * stride + nn]; }
  return b;
}

// ---------------- f32 -> f16 pack ----------------

__global__ void pack_f16_kernel(const float* __restrict__ in, _Float16* __restrict__ out, long n) {
  long i = (long)blockIdx.x * blockDim.x + threadIdx.x;
  const long stride = (long)gridDim.x * blockDim.x;
  for (; i < n; i += stride) out[i] = (_Float16)in[i];
}

// ---------------- GEMM: out[M,2048] = A[M,2048] @ W[2048,2048]^T + bias, masked ----------------
// Block tile 64x128, 8 waves in 2x4, each wave a 32x32 (2x2 WMMA) sub-tile.
// A/B k-slabs double-buffered in LDS, filled by TDM issued from wave 0.

struct GemmSmem {
  _Float16 lA[2][64][32];
  _Float16 lB[2][128][32];
};

__device__ __forceinline__ void gemm_body(GemmSmem& sm,
    const _Float16* __restrict__ A, const _Float16* __restrict__ W,
    const float* __restrict__ bias, const unsigned char* __restrict__ mask,
    float* __restrict__ outp)
{
  const int lane = threadIdx.x & 31;
  const int wave = threadIdx.x >> 5;
  const int wr = wave & 1;          // 2 row groups of 32
  const int wc = wave >> 1;         // 4 col groups of 32
  const int m0 = blockIdx.y * 64;
  const int n0 = blockIdx.x * 128;

  const _Float16* Abase = A + (size_t)m0 * CC;
  const _Float16* Wbase = W + (size_t)n0 * CC;

  v8f acc[2][2] = {};

  if (wave == 0) {  // prologue: DMA first k-slab (TENSORcnt += 2)
    tdm_load_2d(lds_off_u32(&sm.lA[0][0][0]), Abase, 32, 64, CC);
    tdm_load_2d(lds_off_u32(&sm.lB[0][0][0]), Wbase, 32, 128, CC);
  }

  for (int k0 = 0; k0 < CC; k0 += 32) {
    const int cur = (k0 >> 5) & 1;
    if (wave == 0) {
      if (k0 + 32 < CC) {  // DMA next slab into the other buffer; stays in flight
        tdm_load_2d(lds_off_u32(&sm.lA[cur ^ 1][0][0]), Abase + k0 + 32, 32, 64, CC);
        tdm_load_2d(lds_off_u32(&sm.lB[cur ^ 1][0][0]), Wbase + k0 + 32, 32, 128, CC);
        __builtin_amdgcn_s_wait_tensorcnt((short)2);   // current slab complete
      } else {
        __builtin_amdgcn_s_wait_tensorcnt((short)0);
      }
    }
    __syncthreads();

    const v16h a0 = frag_a_lds(&sm.lA[cur][wr * 32 +  0][0], 32, lane);
    const v16h a1 = frag_a_lds(&sm.lA[cur][wr * 32 + 16][0], 32, lane);
    const v16h b0 = frag_b_lds(&sm.lB[cur][wc * 32 +  0][0], 32, lane);
    const v16h b1 = frag_b_lds(&sm.lB[cur][wc * 32 + 16][0], 32, lane);
    acc[0][0] = __builtin_amdgcn_wmma_f32_16x16x32_f16(false, a0, false, b0, (short)0, acc[0][0], false, false);
    acc[0][1] = __builtin_amdgcn_wmma_f32_16x16x32_f16(false, a0, false, b1, (short)0, acc[0][1], false, false);
    acc[1][0] = __builtin_amdgcn_wmma_f32_16x16x32_f16(false, a1, false, b0, (short)0, acc[1][0], false, false);
    acc[1][1] = __builtin_amdgcn_wmma_f32_16x16x32_f16(false, a1, false, b1, (short)0, acc[1][1], false, false);
    __syncthreads();
  }

  // C/D layout: lane&15 = N, (lane>>4)*8+r = M
  const int nn = lane & 15;
  const int mb = (lane >> 4) << 3;
#pragma unroll
  for (int j = 0; j < 2; ++j) {
    const int col = n0 + wc * 32 + j * 16 + nn;
    const float bv = bias[col];
#pragma unroll
    for (int i = 0; i < 2; ++i) {
#pragma unroll
      for (int r = 0; r < 8; ++r) {
        const int row = m0 + wr * 32 + i * 16 + mb + r;
        float v = acc[i][j][r] + bv;
        if (!mask[row]) v = 0.f;
        outp[(size_t)row * CC + col] = v;
      }
    }
  }
}

__global__ __launch_bounds__(256) void gemm_qkv_kernel(
    const _Float16* __restrict__ xh,
    const _Float16* __restrict__ wqh, const _Float16* __restrict__ wkh, const _Float16* __restrict__ wvh,
    const float* __restrict__ bq, const float* __restrict__ bk, const float* __restrict__ bv,
    const unsigned char* __restrict__ mask,
    float* __restrict__ qf, float* __restrict__ kf, float* __restrict__ vf)
{
  __shared__ GemmSmem sm;
  const _Float16* W = (blockIdx.z == 0) ? wqh : (blockIdx.z == 1) ? wkh : wvh;
  const float* bias = (blockIdx.z == 0) ? bq : (blockIdx.z == 1) ? bk : bv;
  float* outp       = (blockIdx.z == 0) ? qf : (blockIdx.z == 1) ? kf : vf;
  gemm_body(sm, xh, W, bias, mask, outp);
}

__global__ __launch_bounds__(256) void gemm_out_kernel(
    const _Float16* __restrict__ ah, const _Float16* __restrict__ woh,
    const float* __restrict__ bo, const unsigned char* __restrict__ mask,
    float* __restrict__ outp)
{
  __shared__ GemmSmem sm;
  gemm_body(sm, ah, woh, bo, mask, outp);
}

// ---------------- RMSNorm (over C=2048) + volumetric RoPE, f32 in -> f16 out ----------------
// t_dim=22, s_dim=21; s -> (f = s>>8, h = (s>>4)&15, w = s&15); freqs_cs[1024][64][2]

__global__ __launch_bounds__(256) void rmsnorm_rope_kernel(
    const float* __restrict__ in, const float* __restrict__ g,
    const float* __restrict__ freqs, _Float16* __restrict__ outh)
{
  __shared__ float red[256];
  __shared__ float rowrms;
  const int row = blockIdx.x;        // 0..4095
  const int t = threadIdx.x;
  const float* p = in + (size_t)row * CC;

  float vals[8];
  float ss = 0.f;
#pragma unroll
  for (int i = 0; i < 8; ++i) { const float x = p[t * 8 + i]; vals[i] = x; ss += x * x; }
  red[t] = ss;
  __syncthreads();
  for (int off = 128; off > 0; off >>= 1) {
    if (t < off) red[t] += red[t + off];
    __syncthreads();
  }
  if (t == 0) rowrms = rsqrtf(red[0] * (1.0f / (float)CC) + 1e-6f);
  __syncthreads();
  const float rms = rowrms;

  const int s = row & (SS - 1);
  const int f = s >> 8, h = (s >> 4) & 15, w = s & 15;

  _Float16 o[8];
#pragma unroll
  for (int pr = 0; pr < 4; ++pr) {
    const int c = t * 8 + pr * 2;
    const int d = c & (HD - 1);
    const int j = d >> 1;                              // pair index in [0,64)
    const int prow = (j < 22) ? f : (j < 43) ? h : w;  // t/h/w angle source
    const float cth = freqs[prow * 128 + j * 2 + 0];
    const float sth = freqs[prow * 128 + j * 2 + 1];
    const float xr = vals[pr * 2]     * rms * g[c];
    const float xi = vals[pr * 2 + 1] * rms * g[c + 1];
    o[pr * 2]     = (_Float16)(xr * cth - xi * sth);
    o[pr * 2 + 1] = (_Float16)(xr * sth + xi * cth);
  }
  _Float16* q = outh + (size_t)row * CC + t * 8;
#pragma unroll
  for (int i = 0; i < 8; ++i) q[i] = o[i];
}

// ---------------- Flash attention: per (b, head), 128 q-rows/block, 8 waves x 16 rows ----------------
// K/V 32-row tiles double-buffered in LDS via TDM from wave 0.

__global__ __launch_bounds__(256) void flash_attn_kernel(
    const _Float16* __restrict__ qhp, const _Float16* __restrict__ khp,
    const _Float16* __restrict__ vhp, const unsigned char* __restrict__ mask,
    _Float16* __restrict__ outh)
{
  __shared__ _Float16 lK[2][32][128];
  __shared__ _Float16 lV[2][32][128];
  __shared__ _Float16 lP[8][16][40];      // per-wave P staging (16 rows x 32 k, pad 8)

  const int lane = threadIdx.x & 31;
  const int wave = threadIdx.x >> 5;
  const int b = blockIdx.y >> 4;
  const int n = blockIdx.y & 15;
  const int s0 = blockIdx.x * 128;
  const int nn = lane & 15;
  const int mb = (lane >> 4) << 3;

  const _Float16* kbase = khp + ((size_t)(b * SS)) * CC + n * HD;
  const _Float16* vbase = vhp + ((size_t)(b * SS)) * CC + n * HD;

  // Q fragments held in registers: 16 rows x 128 d = 4 A-fragments
  v16h qreg[4];
  {
    const int r  = lane & 15;
    const int kb = (lane >> 4) << 3;
    const _Float16* qp = qhp + ((size_t)(b * SS + s0 + wave * 16 + r)) * CC + n * HD;
#pragma unroll
    for (int d0 = 0; d0 < 4; ++d0)
#pragma unroll
      for (int i = 0; i < 8; ++i) {
        qreg[d0][i]     = qp[d0 * 32 + kb + i];
        qreg[d0][i + 8] = qp[d0 * 32 + 16 + kb + i];
      }
  }

  if (wave == 0) {  // prologue: DMA first K/V tile
    tdm_load_2d(lds_off_u32(&lK[0][0][0]), kbase, 128, 32, CC);
    tdm_load_2d(lds_off_u32(&lV[0][0][0]), vbase, 128, 32, CC);
  }

  v8f O[8] = {};                 // 16 x 128 f32 output accumulator
  float m_[8], l_[8];
#pragma unroll
  for (int r = 0; r < 8; ++r) { m_[r] = -1e30f; l_[r] = 0.f; }

  const float scale = 0.08838834764831845f;   // 1/sqrt(128)

  for (int k0 = 0; k0 < SS; k0 += 32) {
    const int cur = (k0 >> 5) & 1;
    if (wave == 0) {
      if (k0 + 32 < SS) {
        tdm_load_2d(lds_off_u32(&lK[cur ^ 1][0][0]), kbase + (size_t)(k0 + 32) * CC, 128, 32, CC);
        tdm_load_2d(lds_off_u32(&lV[cur ^ 1][0][0]), vbase + (size_t)(k0 + 32) * CC, 128, 32, CC);
        __builtin_amdgcn_s_wait_tensorcnt((short)2);
      } else {
        __builtin_amdgcn_s_wait_tensorcnt((short)0);
      }
    }
    __syncthreads();

    // scores = Q @ K^T : two 16x16 tiles, K-dim = 128 in 4 steps
    v8f sc[2] = {};
#pragma unroll
    for (int j = 0; j < 2; ++j) {
#pragma unroll
      for (int d0 = 0; d0 < 4; ++d0) {
        const v16h bf = frag_b_lds(&lK[cur][j * 16][d0 * 32], 128, lane);
        sc[j] = __builtin_amdgcn_wmma_f32_16x16x32_f16(false, qreg[d0], false, bf, (short)0, sc[j], false, false);
      }
      const int kcol = k0 + j * 16 + nn;
      const float biasv = mask[b * SS + kcol] ? 0.f : -1e9f;
#pragma unroll
      for (int r = 0; r < 8; ++r) sc[j][r] = sc[j][r] * scale + biasv;
    }

    // online softmax: row reductions across the 16-lane column groups
    float rm[8];
#pragma unroll
    for (int r = 0; r < 8; ++r) rm[r] = fmaxf(sc[0][r], sc[1][r]);
#pragma unroll
    for (int off = 1; off < 16; off <<= 1)
#pragma unroll
      for (int r = 0; r < 8; ++r) rm[r] = fmaxf(rm[r], __shfl_xor(rm[r], off, 32));

    float alpha[8], rs[8];
#pragma unroll
    for (int r = 0; r < 8; ++r) {
      const float mn = fmaxf(m_[r], rm[r]);
      alpha[r] = __expf(m_[r] - mn);
      m_[r] = mn;
      rs[r] = 0.f;
    }
#pragma unroll
    for (int j = 0; j < 2; ++j)
#pragma unroll
      for (int r = 0; r < 8; ++r) {
        const float pv = __expf(sc[j][r] - m_[r]);
        sc[j][r] = pv;
        rs[r] += pv;
      }
#pragma unroll
    for (int off = 1; off < 16; off <<= 1)
#pragma unroll
      for (int r = 0; r < 8; ++r) rs[r] += __shfl_xor(rs[r], off, 32);
#pragma unroll
    for (int r = 0; r < 8; ++r) l_[r] = l_[r] * alpha[r] + rs[r];
#pragma unroll
    for (int t8 = 0; t8 < 8; ++t8)
#pragma unroll
      for (int r = 0; r < 8; ++r) O[t8][r] *= alpha[r];

    // stage P through per-wave LDS to reshape acc-layout -> A-fragment layout
#pragma unroll
    for (int j = 0; j < 2; ++j)
#pragma unroll
      for (int r = 0; r < 8; ++r)
        lP[wave][mb + r][j * 16 + nn] = (_Float16)sc[j][r];

    const v16h pf = frag_a_lds(&lP[wave][0][0], 40, lane);
#pragma unroll
    for (int dt = 0; dt < 8; ++dt) {
      const v16h vfr = frag_b_rowmajor(&lV[cur][0][dt * 16], 128, lane);
      O[dt] = __builtin_amdgcn_wmma_f32_16x16x32_f16(false, pf, false, vfr, (short)0, O[dt], false, false);
    }
    __syncthreads();
  }

#pragma unroll
  for (int r = 0; r < 8; ++r) l_[r] = 1.f / fmaxf(l_[r], 1e-20f);
#pragma unroll
  for (int dt = 0; dt < 8; ++dt)
#pragma unroll
    for (int r = 0; r < 8; ++r) {
      const int qrow = s0 + wave * 16 + mb + r;
      outh[((size_t)(b * SS + qrow)) * CC + n * HD + dt * 16 + nn] = (_Float16)(O[dt][r] * l_[r]);
    }
}

// ---------------- host-side orchestration ----------------

extern "C" void kernel_launch(void* const* d_in, const int* in_sizes, int n_in,
                              void* d_out, int out_size, void* d_ws, size_t ws_size,
                              hipStream_t stream) {
  (void)in_sizes; (void)n_in; (void)out_size; (void)ws_size;
  const float* x     = (const float*)d_in[0];
  const float* freqs = (const float*)d_in[1];
  const float* wq    = (const float*)d_in[2];
  const float* bq    = (const float*)d_in[3];
  const float* wk    = (const float*)d_in[4];
  const float* bk    = (const float*)d_in[5];
  const float* wv    = (const float*)d_in[6];
  const float* bv    = (const float*)d_in[7];
  const float* wo    = (const float*)d_in[8];
  const float* bo    = (const float*)d_in[9];
  const float* gq    = (const float*)d_in[10];
  const float* gk    = (const float*)d_in[11];
  const unsigned char* mask = (const unsigned char*)d_in[12];
  float* out = (float*)d_out;

  size_t off = 0;
  auto alloc = [&](size_t bytes) -> void* {
    void* p = (char*)d_ws + off;
    off += (bytes + 255) & ~(size_t)255;
    return p;
  };
  _Float16* xh  = (_Float16*)alloc((size_t)MM * CC * 2);
  _Float16* wqh = (_Float16*)alloc((size_t)CC * CC * 2);
  _Float16* wkh = (_Float16*)alloc((size_t)CC * CC * 2);
  _Float16* wvh = (_Float16*)alloc((size_t)CC * CC * 2);
  _Float16* woh = (_Float16*)alloc((size_t)CC * CC * 2);
  float*    qf  = (float*)   alloc((size_t)MM * CC * 4);
  float*    kf  = (float*)   alloc((size_t)MM * CC * 4);
  float*    vf  = (float*)   alloc((size_t)MM * CC * 4);
  _Float16* qh  = (_Float16*)alloc((size_t)MM * CC * 2);
  _Float16* kh  = (_Float16*)alloc((size_t)MM * CC * 2);
  _Float16* vh  = (_Float16*)alloc((size_t)MM * CC * 2);
  _Float16* ah  = (_Float16*)alloc((size_t)MM * CC * 2);

  const dim3 blk(256);

  pack_f16_kernel<<<512, blk, 0, stream>>>(x,  xh,  (long)MM * CC);
  pack_f16_kernel<<<512, blk, 0, stream>>>(wq, wqh, (long)CC * CC);
  pack_f16_kernel<<<512, blk, 0, stream>>>(wk, wkh, (long)CC * CC);
  pack_f16_kernel<<<512, blk, 0, stream>>>(wv, wvh, (long)CC * CC);
  pack_f16_kernel<<<512, blk, 0, stream>>>(wo, woh, (long)CC * CC);

  const dim3 ggrid(CC / 128, MM / 64, 3);
  gemm_qkv_kernel<<<ggrid, blk, 0, stream>>>(xh, wqh, wkh, wvh, bq, bk, bv, mask, qf, kf, vf);

  rmsnorm_rope_kernel<<<MM, blk, 0, stream>>>(qf, gq, freqs, qh);
  rmsnorm_rope_kernel<<<MM, blk, 0, stream>>>(kf, gk, freqs, kh);
  pack_f16_kernel<<<512, blk, 0, stream>>>(vf, vh, (long)MM * CC);

  const dim3 fgrid(SS / 128, 32);
  flash_attn_kernel<<<fgrid, blk, 0, stream>>>(qh, kh, vh, mask, ah);

  const dim3 ogrid(CC / 128, MM / 64, 1);
  gemm_out_kernel<<<ogrid, blk, 0, stream>>>(ah, woh, bo, mask, out);
}
